// GCN_48077863911469
// MI455X (gfx1250) — matrix operand order repaired and checked by reading, not compiled, not run on hardware
//
#include <hip/hip_runtime.h>

typedef __bf16 bf16_t;
typedef __attribute__((ext_vector_type(16))) __bf16 v16bf;
typedef __attribute__((ext_vector_type(8)))  float  v8f;

__device__ __forceinline__ bf16_t f2bf(float f) {
  // round-to-nearest-even float32 -> bf16
  unsigned u = __builtin_bit_cast(unsigned, f);
  u += 0x7FFFu + ((u >> 16) & 1u);
  unsigned short h = (unsigned short)(u >> 16);
  return __builtin_bit_cast(bf16_t, h);
}

// ---------------- conversion / transpose ----------------

// dst[i] = bf16( relu? max(src,0) : src )
__global__ void k_cvt_bf16(const float* __restrict__ src, bf16_t* __restrict__ dst,
                           int n, int do_relu) {
  int i = blockIdx.x * blockDim.x + threadIdx.x;
  if (i >= n) return;
  float v = src[i];
  if (do_relu) v = fmaxf(v, 0.0f);
  dst[i] = f2bf(v);
}

// src is [K][Nc] fp32 row-major; dst is [Nc][K] bf16 (transposed)
__global__ void k_transpose_bf16(const float* __restrict__ src, bf16_t* __restrict__ dst,
                                 int K, int Nc) {
  int i = blockIdx.x * blockDim.x + threadIdx.x;
  if (i >= K * Nc) return;
  int k = i / Nc, n = i % Nc;
  dst[(size_t)n * K + k] = f2bf(src[i]);
}

// ---------------- graph normalization ----------------

__global__ void k_init_deg(float* __restrict__ deg, int n) {
  int i = blockIdx.x * blockDim.x + threadIdx.x;
  if (i < n) deg[i] = 1.0f;            // self-loop weight
}

__global__ void k_accum_deg(const int* __restrict__ col, const float* __restrict__ ew,
                            float* __restrict__ deg, int E) {
  int e = blockIdx.x * blockDim.x + threadIdx.x;
  if (e < E) atomicAdd(&deg[col[e]], ew[e]);
}

__global__ void k_dinv(const float* __restrict__ deg, float* __restrict__ dinv, int n) {
  int i = blockIdx.x * blockDim.x + threadIdx.x;
  if (i < n) {
    float d = deg[i];
    dinv[i] = (d > 0.0f) ? rsqrtf(d) : 0.0f;
  }
}

// agg[i][f] = dinv[i]^2 * h[i][f] + bias[f]   (self-loop contribution + bias)
__global__ void k_agg_init(const float* __restrict__ h, const float* __restrict__ dinv,
                           const float* __restrict__ bias, float* __restrict__ agg,
                           int N, int F) {
  int i = blockIdx.x * blockDim.x + threadIdx.x;
  if (i >= N * F) return;
  int node = i / F, f = i % F;
  float di = dinv[node];
  agg[i] = di * di * h[i] + bias[f];
}

// agg[col[e]][f..f+3] += norm_e * h[row[e]][f..f+3]
__global__ void k_agg_scatter(const int* __restrict__ row, const int* __restrict__ col,
                              const float* __restrict__ ew, const float* __restrict__ dinv,
                              const float* __restrict__ h, float* __restrict__ agg,
                              int E, int F) {
  int chunks = F >> 2;
  int idx = blockIdx.x * blockDim.x + threadIdx.x;
  if (idx >= E * chunks) return;
  int e = idx / chunks;
  int f4 = (idx % chunks) << 2;
  int r = row[e], c = col[e];
  float norm = dinv[r] * ew[e] * dinv[c];
  const float4 hv = *(const float4*)(h + (size_t)r * F + f4);
  float* dst = agg + (size_t)c * F + f4;
  atomicAdd(dst + 0, norm * hv.x);
  atomicAdd(dst + 1, norm * hv.y);
  atomicAdd(dst + 2, norm * hv.z);
  atomicAdd(dst + 3, norm * hv.w);
}

// ---------------- WMMA GEMM:  C[M][N] = A_bf16[M][K] @ Bt_bf16[N][K]^T (+bias) ----------------
// One wave per 16x16 C tile; K multiple of 32; M,N multiples of 16.

__global__ void k_wmma_gemm(const bf16_t* __restrict__ A, const bf16_t* __restrict__ Bt,
                            float* __restrict__ C, const float* __restrict__ bias,
                            int M, int N, int K) {
  int wave = (int)((blockIdx.x * blockDim.x + threadIdx.x) >> 5);
  int lane = threadIdx.x & 31;
  int tilesN = N >> 4;
  int tilesM = M >> 4;
  int tm = wave / tilesN;
  int tn = wave % tilesN;
  if (tm >= tilesM) return;                 // wave-uniform: EXEC stays all-ones

  int hi = lane >> 4;                        // 0 or 1 (lane half)
  int lo = lane & 15;

  const bf16_t* arow = A  + (size_t)(tm * 16 + lo) * K;   // A fragment: row m = lo
  const bf16_t* brow = Bt + (size_t)(tn * 16 + lo) * K;   // B fragment: col n = lo

  v8f acc = {};
  for (int k0 = 0; k0 < K; k0 += 32) {
    // A 16-bit layout (ISA 7.12.2): lane half 'hi' holds k = k0+8*hi+(0..7)
    // and k0+16+8*hi+(0..7)  -> two 16B chunks
    const bf16_t* ap = arow + k0 + 8 * hi;
    // B 16-bit layout: same as A, lane half 'hi' holds k = k0+8*hi+(0..7)
    // and k0+16+8*hi+(0..7) for its column  -> two 16B chunks
    const bf16_t* bp = brow + k0 + 8 * hi;
    v16bf a, b;
#pragma unroll
    for (int t = 0; t < 8; ++t) {
      a[t]     = ap[t];
      a[t + 8] = ap[16 + t];
    }
#pragma unroll
    for (int t = 0; t < 8; ++t) {
      b[t]     = bp[t];
      b[t + 8] = bp[16 + t];
    }
    acc = __builtin_amdgcn_wmma_f32_16x16x32_bf16(
        /*neg_a=*/false, a, /*neg_b=*/false, b,
        /*c_mod=*/(short)0, acc, /*reuse_a=*/false, /*reuse_b=*/false);
  }

  // C/D layout: VGPR r -> row (r + 8*hi), col lo
  int ncol = tn * 16 + lo;
  float badd = bias ? bias[ncol] : 0.0f;
#pragma unroll
  for (int r = 0; r < 8; ++r) {
    int mrow = tm * 16 + r + 8 * hi;
    C[(size_t)mrow * N + ncol] = acc[r] + badd;
  }
}

// ---------------- decoder:  adj[i][j] = sigmoid( sum_k relu(A[i,k]+B[j,k]) * w2[k] + db2 ) ----

__global__ __launch_bounds__(256)
void k_decoder(const float* __restrict__ Ab,   // [N][128], db1 already folded in
               const float* __restrict__ Bb,   // [N][128]
               const float* __restrict__ dW2,  // [128]
               const float* __restrict__ db2,  // [1]
               float* __restrict__ adj, int N) {
  __shared__ float As[16][128];
  __shared__ float Bs[16][128];
  __shared__ float Ws[128];
  int tj = threadIdx.x, ti = threadIdx.y;     // 16x16 block
  int t = ti * 16 + tj;
  int i0 = blockIdx.y * 16, j0 = blockIdx.x * 16;

  for (int s = t; s < 16 * 128; s += 256) {
    int r = s >> 7, c = s & 127;
    As[r][c] = Ab[(size_t)(i0 + r) * 128 + c];
    Bs[r][c] = Bb[(size_t)(j0 + r) * 128 + c];
  }
  if (t < 128) Ws[t] = dW2[t];
  __syncthreads();

  const float* a = As[ti];
  const float* b = Bs[tj];
  float acc = 0.0f;
#pragma unroll 8
  for (int k = 0; k < 128; ++k)
    acc += fmaxf(a[k] + b[k], 0.0f) * Ws[k];

  float s = acc + db2[0];
  adj[(size_t)(i0 + ti) * N + (j0 + tj)] = 1.0f / (1.0f + __expf(-s));
}

// ---------------- host side ----------------

static inline int cdiv(int a, int b) { return (a + b - 1) / b; }

extern "C" void kernel_launch(void* const* d_in, const int* in_sizes, int n_in,
                              void* d_out, int out_size, void* d_ws, size_t ws_size,
                              hipStream_t stream) {
  const float* x   = (const float*)d_in[0];
  const int*   ei  = (const int*)  d_in[1];
  const float* ew  = (const float*)d_in[2];
  const float* W1  = (const float*)d_in[3];
  const float* b1  = (const float*)d_in[4];
  const float* W2  = (const float*)d_in[5];
  const float* b2  = (const float*)d_in[6];
  const float* dW1 = (const float*)d_in[7];
  const float* db1 = (const float*)d_in[8];
  const float* dW2 = (const float*)d_in[9];
  const float* db2 = (const float*)d_in[10];

  const int N = 1024, F = 256, Hd = 256, D = 64, Mh = 128;
  const int E = in_sizes[1] / 2;
  const int* row = ei;
  const int* col = ei + E;

  float* zout = (float*)d_out;              // [N][D]
  float* adj  = (float*)d_out + (size_t)N * D;  // [N][N]

  // bump allocator over d_ws
  char* p = (char*)d_ws;
  auto alloc = [&](size_t bytes) -> void* {
    void* r = (void*)p;
    p += (bytes + 255) & ~(size_t)255;
    return r;
  };
  float*  deg  = (float*) alloc((size_t)N * 4);
  float*  dinv = (float*) alloc((size_t)N * 4);
  bf16_t* xb   = (bf16_t*)alloc((size_t)N * F * 2);
  bf16_t* W1t  = (bf16_t*)alloc((size_t)Hd * F * 2);   // [Hd][F]
  bf16_t* W2t  = (bf16_t*)alloc((size_t)D * Hd * 2);   // [D][Hd]
  bf16_t* dAt  = (bf16_t*)alloc((size_t)Mh * D * 2);   // [Mh][D]  (dW1[:64]^T)
  bf16_t* dBt  = (bf16_t*)alloc((size_t)Mh * D * 2);   // [Mh][D]  (dW1[64:]^T)
  float*  h0   = (float*) alloc((size_t)N * Hd * 4);   // x @ W1
  float*  agg1 = (float*) alloc((size_t)N * Hd * 4);   // layer-1 aggregate
  bf16_t* hbf  = (bf16_t*)alloc((size_t)N * Hd * 2);   // relu(agg1) bf16
  float*  hz   = (float*) alloc((size_t)N * D * 4);    // h @ W2
  bf16_t* zbf  = (bf16_t*)alloc((size_t)N * D * 2);
  float*  Abuf = (float*) alloc((size_t)N * Mh * 4);   // z@dW1[:64] + db1
  float*  Bbuf = (float*) alloc((size_t)N * Mh * 4);   // z@dW1[64:]

  const int T = 256;

  // --- precision conversion + weight transposes ---
  k_cvt_bf16     <<<cdiv(N * F, T), T, 0, stream>>>(x, xb, N * F, 0);
  k_transpose_bf16<<<cdiv(F * Hd, T), T, 0, stream>>>(W1, W1t, F, Hd);
  k_transpose_bf16<<<cdiv(Hd * D, T), T, 0, stream>>>(W2, W2t, Hd, D);
  k_transpose_bf16<<<cdiv(D * Mh, T), T, 0, stream>>>(dW1,            dAt, D, Mh);
  k_transpose_bf16<<<cdiv(D * Mh, T), T, 0, stream>>>(dW1 + (size_t)D * Mh, dBt, D, Mh);

  // --- symmetric normalization ---
  k_init_deg <<<cdiv(N, T), T, 0, stream>>>(deg, N);
  k_accum_deg<<<cdiv(E, T), T, 0, stream>>>(col, ew, deg, E);
  k_dinv     <<<cdiv(N, T), T, 0, stream>>>(deg, dinv, N);

  // --- GCN layer 1: h = relu(aggregate(x @ W1) + b1) ---
  {
    int tiles = (N / 16) * (Hd / 16);
    k_wmma_gemm<<<cdiv(tiles * 32, T), T, 0, stream>>>(xb, W1t, h0, nullptr, N, Hd, F);
  }
  k_agg_init   <<<cdiv(N * Hd, T), T, 0, stream>>>(h0, dinv, b1, agg1, N, Hd);
  k_agg_scatter<<<cdiv(E * (Hd / 4), T), T, 0, stream>>>(row, col, ew, dinv, h0, agg1, E, Hd);
  k_cvt_bf16   <<<cdiv(N * Hd, T), T, 0, stream>>>(agg1, hbf, N * Hd, 1);

  // --- GCN layer 2: z = aggregate(h @ W2) + b2  (written straight into d_out) ---
  {
    int tiles = (N / 16) * (D / 16);
    k_wmma_gemm<<<cdiv(tiles * 32, T), T, 0, stream>>>(hbf, W2t, hz, nullptr, N, D, Hd);
  }
  k_agg_init   <<<cdiv(N * D, T), T, 0, stream>>>(hz, dinv, b2, zout, N, D);
  k_agg_scatter<<<cdiv(E * (D / 4), T), T, 0, stream>>>(row, col, ew, dinv, hz, zout, E, D);
  k_cvt_bf16   <<<cdiv(N * D, T), T, 0, stream>>>(zout, zbf, N * D, 0);

  // --- decoder projections: A = z@dW1[:64] + db1 ; B = z@dW1[64:] ---
  {
    int tiles = (N / 16) * (Mh / 16);
    k_wmma_gemm<<<cdiv(tiles * 32, T), T, 0, stream>>>(zbf, dAt, Abuf, db1,    N, Mh, D);
    k_wmma_gemm<<<cdiv(tiles * 32, T), T, 0, stream>>>(zbf, dBt, Bbuf, nullptr, N, Mh, D);
  }

  // --- all-pairs decoder ---
  {
    dim3 grid(N / 16, N / 16), blk(16, 16);
    k_decoder<<<grid, blk, 0, stream>>>(Abuf, Bbuf, dW2, db2, adj, N);
  }
}
